// SAHead_14328010900300
// MI455X (gfx1250) — matrix operand order repaired and checked by reading, not compiled, not run on hardware
//
#include <hip/hip_runtime.h>

#define BB 2
#define CC 64
#define CE 96
#define DK 64
#define DV 64
#define HH 96
#define WW 96
#define LL (HH * WW)   // 9216 spatial positions

typedef _Float16 h8   __attribute__((ext_vector_type(8)));
typedef _Float16 h16  __attribute__((ext_vector_type(16)));
typedef float    f8   __attribute__((ext_vector_type(8)));
typedef unsigned int  u32x4 __attribute__((ext_vector_type(4)));
typedef int           i32x4 __attribute__((ext_vector_type(4)));
typedef int           i32x8 __attribute__((ext_vector_type(8)));

__device__ __forceinline__ f8 wmma_f16(h16 a, h16 b, f8 c) {
    // D = A(16x32 f16) * B(32x16 f16) + C(16x16 f32)
    return __builtin_amdgcn_wmma_f32_16x16x32_f16(
        /*neg_a=*/false, a, /*neg_b=*/false, b,
        /*c_mod=*/(short)0, c, /*reuse_a=*/false, /*reuse_b=*/false);
}

// LDS byte offset: per ISA aperture mapping, the LDS address is the low 32 bits of the flat address
__device__ __forceinline__ unsigned int lds_off(const void* p) {
    return (unsigned int)(uintptr_t)p;
}

// ---------------------------------------------------------------------------
// TDM 2D tile load Global->LDS (CDNA5 ISA D# layout).
// data_size = 4B units. tensor dims == tile dims so no OOB zero-fill.
// ---------------------------------------------------------------------------
__device__ __forceinline__ void tdm_load_2d(
    unsigned int ldsByteOff, const void* gptr,
    unsigned int tileD0_du,   // tile width in dwords (16-bit field)
    unsigned int tileD1,      // tile rows        (16-bit field)
    unsigned long long strideDu) // row stride in dwords (48-bit field)
{
    unsigned long long ga = (unsigned long long)(uintptr_t)gptr;
    u32x4 g0;
    g0[0] = 1u;                                   // count=1, user descriptor
    g0[1] = ldsByteOff;                           // lds_addr [63:32]
    g0[2] = (unsigned int)(ga & 0xFFFFFFFFu);     // global_addr lo
    g0[3] = (unsigned int)((ga >> 32) & 0x01FFFFFFu) | 0x80000000u; // type=2
    i32x8 g1;
    unsigned int td0 = tileD0_du, td1 = tileD1;   // tensor dims = tile dims
    g1[0] = (int)(2u << 16);                      // data_size = 4 bytes
    g1[1] = (int)((td0 & 0xFFFFu) << 16);         // tensor_dim0[15:0]
    g1[2] = (int)(((td0 >> 16) & 0xFFFFu) | ((td1 & 0xFFFFu) << 16));
    g1[3] = (int)(((td1 >> 16) & 0xFFFFu) | ((tileD0_du & 0xFFFFu) << 16)); // tile_dim0
    g1[4] = (int)(tileD1 & 0xFFFFu);              // tile_dim1 (tile_dim2=0)
    g1[5] = (int)(unsigned int)(strideDu & 0xFFFFFFFFu);   // dim0_stride lo32
    g1[6] = (int)(unsigned int)((strideDu >> 32) & 0xFFFFu); // dim0_stride hi16
    g1[7] = 0;
    i32x4 z4 = {0, 0, 0, 0};
#if defined(__clang_major__) && (__clang_major__ >= 23)
    i32x8 z8 = {0, 0, 0, 0, 0, 0, 0, 0};
    __builtin_amdgcn_tensor_load_to_lds(g0, g1, z4, z4, z8, 0);
#else
    __builtin_amdgcn_tensor_load_to_lds(g0, g1, z4, z4, 0);
#endif
}

// ---------------------------------------------------------------------------
// Kernel 1: pointwise projections.
//   Q,K -> f32 (unnormalized, position-major [B][L][64])
//   V   -> f16 (channel-major  [B][64][L])
// ---------------------------------------------------------------------------
__global__ __launch_bounds__(256) void proj_kernel(
    const float* __restrict__ x, const float* __restrict__ x_enc,
    const float* __restrict__ Wq, const float* __restrict__ bq,
    const float* __restrict__ Wk, const float* __restrict__ bk,
    const float* __restrict__ Wv, const float* __restrict__ bv,
    float* __restrict__ qF, float* __restrict__ kF, _Float16* __restrict__ vD)
{
    int t = blockIdx.x * 256 + threadIdx.x;
    if (t >= BB * LL) return;
    int b = t / LL;
    int l = t - b * LL;

    float xe[CE];
#pragma unroll
    for (int c = 0; c < CE; ++c)
        xe[c] = x_enc[((size_t)b * CE + c) * LL + l];

    float* qrow = qF + (size_t)t * DK;
    for (int o = 0; o < DK; ++o) {
        float a = bq[o];
#pragma unroll
        for (int c = 0; c < CE; ++c) a += Wq[o * CE + c] * xe[c];
        qrow[o] = a;
    }
    float* krow = kF + (size_t)t * DK;
    for (int o = 0; o < DK; ++o) {
        float a = bk[o];
#pragma unroll
        for (int c = 0; c < CE; ++c) a += Wk[o * CE + c] * xe[c];
        krow[o] = a;
    }

    float xc[CC];
#pragma unroll
    for (int c = 0; c < CC; ++c)
        xc[c] = x[((size_t)b * CC + c) * LL + l];
    for (int o = 0; o < DV; ++o) {
        float a = bv[o];
#pragma unroll
        for (int c = 0; c < CC; ++c) a += Wv[o * CC + c] * xc[c];
        vD[((size_t)b * DV + o) * LL + l] = (_Float16)a;
    }
}

// ---------------------------------------------------------------------------
// Kernel 2: per-position L2 normalization (channel dim, clamp at eps) -> f16
// ---------------------------------------------------------------------------
__global__ __launch_bounds__(256) void norm_kernel(
    const float* __restrict__ F, _Float16* __restrict__ Hout)
{
    int t = blockIdx.x * 256 + threadIdx.x;
    if (t >= BB * LL) return;
    const float* r = F + (size_t)t * DK;
    float ss = 0.0f;
#pragma unroll
    for (int o = 0; o < DK; ++o) { float v = r[o]; ss += v * v; }
    float n = fmaxf(sqrtf(ss), 1e-6f);
    float inv = 1.0f / n;
    _Float16* h = Hout + (size_t)t * DK;
#pragma unroll
    for (int o = 0; o < DK; ++o) h[o] = (_Float16)(r[o] * inv);
}

// ---------------------------------------------------------------------------
// Kernel 3: streaming cosine attention with WMMA + TDM double buffering.
// Block = 256 threads = 8 waves; wave w owns query rows [i0, i0+16).
// Per key tile of 32 (288 iterations):
//   wave0 issues TDM loads of the NEXT K (32x128B) and V (64x64B) tiles,
//   s_wait_tensorcnt keeps one tile in flight; barrier; all waves:
//   - hoist ALL 8 K/V LDS fragments into distinct registers (one DS clause,
//     one wait) so the 4 QK WMMAs run back-to-back and the V loads overlap
//     the softmax VALU work,
//   - exp + P relayout (per-wave LDS scratch) + 4 PV WMMAs.
// ---------------------------------------------------------------------------
__global__ __launch_bounds__(256) void attn_kernel(
    const _Float16* __restrict__ qH, const _Float16* __restrict__ kH,
    const _Float16* __restrict__ vD, float* __restrict__ out)
{
    __shared__ __align__(32) _Float16 Kt[2][32 * 64];   // 2 x 4KB key tiles
    __shared__ __align__(32) _Float16 Vt[2][64 * 32];   // 2 x 4KB value tiles
    __shared__ __align__(32) _Float16 Plds[8 * 16 * 32];// per-wave P scratch

    const int lane = threadIdx.x & 31;
    const int wave = threadIdx.x >> 5;
    const int hi   = lane >> 4;
    const int lo   = lane & 15;
    const int tilesPerBatch = LL / 128;                 // 72
    const int b  = blockIdx.x / tilesPerBatch;
    const int i0 = (blockIdx.x % tilesPerBatch) * 128 + wave * 16;

    const _Float16* kB = kH + (size_t)b * LL * DK;
    const _Float16* vB = vD + (size_t)b * DV * LL;

    // --- Q as two A-fragments (d = 0..31 and d = 32..63) ---
    const _Float16* qrow = qH + ((size_t)b * LL + i0 + lo) * DK;
    h16 aq0, aq1;
    {
        h8 p0 = *(const h8*)(qrow + hi * 16);
        h8 p1 = *(const h8*)(qrow + hi * 16 + 8);
        h8 p2 = *(const h8*)(qrow + 32 + hi * 16);
        h8 p3 = *(const h8*)(qrow + 32 + hi * 16 + 8);
#pragma unroll
        for (int i = 0; i < 8; ++i) {
            aq0[i] = p0[i]; aq0[i + 8] = p1[i];
            aq1[i] = p2[i]; aq1[i + 8] = p3[i];
        }
    }

    f8 O0 = {}, O1 = {}, O2 = {}, O3 = {};
    float lsum[8] = {0.f, 0.f, 0.f, 0.f, 0.f, 0.f, 0.f, 0.f};
    _Float16* Pw = Plds + wave * (16 * 32);

    const int NIT = LL / 32;                            // 288
    if (wave == 0) {
        // prologue: tile 0 -> buffer 0 (K: 32 rows x 32 dwords; V: 64 rows x 16 dwords)
        tdm_load_2d(lds_off(&Kt[0][0]), kB, 32, 32, 32ull);
        tdm_load_2d(lds_off(&Vt[0][0]), vB, 16, 64, (unsigned long long)(LL / 2));
    }

    for (int it = 0; it < NIT; ++it) {
        const int buf = it & 1;
        if (wave == 0) {
            if (it + 1 < NIT) {
                const int jn = (it + 1) * 32;
                tdm_load_2d(lds_off(&Kt[buf ^ 1][0]), kB + (size_t)jn * DK,
                            32, 32, 32ull);
                tdm_load_2d(lds_off(&Vt[buf ^ 1][0]), vB + jn,
                            16, 64, (unsigned long long)(LL / 2));
                __builtin_amdgcn_s_wait_tensorcnt(2);   // current tile done
            } else {
                __builtin_amdgcn_s_wait_tensorcnt(0);
            }
        }
        __syncthreads();                                // tile `it` visible to all

        // ---- hoist all operand fragments out of LDS (distinct registers) ----
        const _Float16* Kb  = &Kt[buf][0];
        const _Float16* Vb  = &Vt[buf][0];
        const _Float16* kr0 = Kb + lo * DK + hi * 16;
        const _Float16* kr1 = Kb + (16 + lo) * DK + hi * 16;
        h16 kb0 = *(const h16*)(kr0);
        h16 kb1 = *(const h16*)(kr0 + 32);
        h16 kb2 = *(const h16*)(kr1);
        h16 kb3 = *(const h16*)(kr1 + 32);
        const _Float16* vr = Vb + lo * 32 + hi * 16;
        h16 vb0 = *(const h16*)(vr);
        h16 vb1 = *(const h16*)(vr + 16 * 32);
        h16 vb2 = *(const h16*)(vr + 32 * 32);
        h16 vb3 = *(const h16*)(vr + 48 * 32);

        // ---- S = Q^T K for 32 keys (interleaved accumulator chains) ----
        f8 S0 = {}, S1 = {};
        S0 = wmma_f16(aq0, kb0, S0);
        S1 = wmma_f16(aq0, kb2, S1);
        S0 = wmma_f16(aq1, kb1, S0);
        S1 = wmma_f16(aq1, kb3, S1);

        // ---- P = exp(S); D-layout -> A-layout via per-wave LDS scratch ----
        __builtin_amdgcn_wave_barrier();
#pragma unroll
        for (int r = 0; r < 8; ++r) {
            float e0 = __expf(S0[r]);
            float e1 = __expf(S1[r]);
            lsum[r] += e0 + e1;
            int row = r + 8 * hi;
            Pw[row * 32 + lo]      = (_Float16)e0;
            Pw[row * 32 + 16 + lo] = (_Float16)e1;
        }
        __builtin_amdgcn_wave_barrier();
        asm volatile("s_wait_dscnt 0" ::: "memory");

        h16 ap;
        {
            h8 p0 = *(const h8*)(Pw + lo * 32 + hi * 16);
            h8 p1 = *(const h8*)(Pw + lo * 32 + hi * 16 + 8);
#pragma unroll
            for (int i = 0; i < 8; ++i) { ap[i] = p0[i]; ap[i + 8] = p1[i]; }
        }
        __builtin_amdgcn_wave_barrier();

        // ---- O += P * V^T (V fragments already resident in registers) ----
        O0 = wmma_f16(ap, vb0, O0);
        O1 = wmma_f16(ap, vb1, O1);
        O2 = wmma_f16(ap, vb2, O2);
        O3 = wmma_f16(ap, vb3, O3);

        __syncthreads();                                // buffer reusable
    }

    // ---- finalize: full row sums via 16-lane butterfly, divide, store ----
#pragma unroll
    for (int r = 0; r < 8; ++r) {
        float t = lsum[r];
        t += __shfl_xor(t, 1, 32);
        t += __shfl_xor(t, 2, 32);
        t += __shfl_xor(t, 4, 32);
        t += __shfl_xor(t, 8, 32);
        float inv = 1.0f / t;
        int i = i0 + r + 8 * hi;   // query index within batch
        out[((size_t)b * DV +      lo) * LL + i] = O0[r] * inv;
        out[((size_t)b * DV + 16 + lo) * LL + i] = O1[r] * inv;
        out[((size_t)b * DV + 32 + lo) * LL + i] = O2[r] * inv;
        out[((size_t)b * DV + 48 + lo) * LL + i] = O3[r] * inv;
    }
}

// ---------------------------------------------------------------------------
extern "C" void kernel_launch(void* const* d_in, const int* in_sizes, int n_in,
                              void* d_out, int out_size, void* d_ws, size_t ws_size,
                              hipStream_t stream)
{
    (void)in_sizes; (void)n_in; (void)out_size; (void)ws_size;
    const float* x     = (const float*)d_in[0];
    const float* x_enc = (const float*)d_in[1];
    const float* Wq    = (const float*)d_in[2];
    const float* bq    = (const float*)d_in[3];
    const float* Wk    = (const float*)d_in[4];
    const float* bk    = (const float*)d_in[5];
    const float* Wv    = (const float*)d_in[6];
    const float* bv    = (const float*)d_in[7];
    float* out = (float*)d_out;

    const size_t nPos = (size_t)BB * LL;          // 18432
    float*     qF = (float*)d_ws;                 // nPos*64 f32
    float*     kF = qF + nPos * DK;               // nPos*64 f32
    _Float16*  qH = (_Float16*)(kF + nPos * DK);  // nPos*64 f16 (normalized)
    _Float16*  kH = qH + nPos * DK;               // nPos*64 f16 (normalized)
    _Float16*  vD = kH + nPos * DK;               // 64*L per batch, f16
    // total ws use: nPos*64*(4+4+2+2+2) = ~16.5 MB

    const int threads = 256;
    const int posBlocks = (int)((nPos + threads - 1) / threads);  // 72

    proj_kernel<<<posBlocks, threads, 0, stream>>>(x, x_enc, Wq, bq, Wk, bk,
                                                   Wv, bv, qF, kF, vD);
    norm_kernel<<<posBlocks, threads, 0, stream>>>(qF, qH);
    norm_kernel<<<posBlocks, threads, 0, stream>>>(kF, kH);

    const int attnBlocks = BB * (LL / 128);       // 144 blocks, 8 waves each
    attn_kernel<<<attnBlocks, threads, 0, stream>>>(qH, kH, vD, out);
}
